// DINONewVq_6073083757238
// MI455X (gfx1250) — matrix operand order repaired and checked by reading, not compiled, not run on hardware
//
#include <hip/hip_runtime.h>
#include <hip/hip_fp16.h>

typedef __attribute__((ext_vector_type(16))) _Float16 v16h;
typedef __attribute__((ext_vector_type(8)))  _Float16 h8;
typedef __attribute__((ext_vector_type(4)))  _Float16 h4;
typedef __attribute__((ext_vector_type(8)))  float    v8f;
typedef __attribute__((ext_vector_type(4)))  float    f4;

#define BB   16
#define CC_  512
#define HWD  784
#define NN   12544      // B*H*W
#define PP   4
#define KK   2048
#define DD   128
#define PK   8192       // P*K
#define NT   98         // N / 128

// e = exp(-dist/T) = exp(-2*(||c||^2 - 2 z.c)) = exp2( C4*acc + CS*||c||^2 )
#define C4_ 5.7707801635558535f     //  4 * log2(e)
#define CS_ (-2.8853900817779268f)  // -2 * log2(e)

union AF { v16h v; h8 h[2]; };

// ---------------- Kernel 1: transpose z (B,C,H,W) -> zf16[p][n][d] (f16, rows contiguous)
__global__ void k_transpose(const float* __restrict__ z, _Float16* __restrict__ zf16) {
    __shared__ float tile[32][33];
    int p = blockIdx.z >> 4, b = blockIdx.z & 15;
    int hw0 = blockIdx.x * 32, di0 = blockIdx.y * 32;
    int tx = threadIdx.x, ty = threadIdx.y;
    const float* src = z + (size_t)(b * CC_ + p * DD) * HWD;
#pragma unroll
    for (int r = 0; r < 4; ++r) {
        int di = di0 + ty + 8 * r;
        int hw = hw0 + tx;
        tile[ty + 8 * r][tx] = (hw < HWD) ? src[(size_t)di * HWD + hw] : 0.f;
    }
    __syncthreads();
    int di = di0 + tx;
#pragma unroll
    for (int r = 0; r < 4; ++r) {
        int hw = hw0 + ty + 8 * r;
        if (hw < HWD) {
            int n = b * HWD + hw;
            zf16[(size_t)(p * NN + n) * DD + di] = (_Float16)tile[tx][ty + 8 * r];
        }
    }
}

// ---------------- Kernel 2: codebooks -> f16 copy + cs[p][k] = -2*log2e*||c_k||^2
__global__ void k_prepcb(const float* __restrict__ cb, _Float16* __restrict__ chalf,
                         float* __restrict__ cs) {
    int wave = threadIdx.x >> 5, lane = threadIdx.x & 31;
    int row = blockIdx.x * 8 + wave;            // p*K + k  (< 8192)
    f4 v = *(const f4*)(cb + (size_t)row * DD + lane * 4);
    h4 hv; hv.x = (_Float16)v.x; hv.y = (_Float16)v.y; hv.z = (_Float16)v.z; hv.w = (_Float16)v.w;
    *(h4*)(chalf + (size_t)row * DD + lane * 4) = hv;
    float s = v.x * v.x + v.y * v.y + v.z * v.z + v.w * v.w;
#pragma unroll
    for (int off = 16; off; off >>= 1) s += __shfl_xor(s, off, 32);
    if (lane == 0) cs[row] = CS_ * s;
}

// ---------------- WMMA helper: acc = A(16x128) x B(cols = codebook rows k0..k0+15)
__device__ __forceinline__ v8f dot128(const AF a[4], const _Float16* __restrict__ brow) {
    v8f acc = {};
#pragma unroll
    for (int c = 0; c < 4; ++c) {
        AF bf;
        bf.h[0] = *(const h8*)(brow + c * 32);
        bf.h[1] = *(const h8*)(brow + c * 32 + 8);
        acc = __builtin_amdgcn_wmma_f32_16x16x32_f16(false, a[c].v, false, bf.v,
                                                     (short)0, acc, false, false);
    }
    return acc;
}

// ---------------- Kernel 3: fused distances + softmax-sum + argmax(e) + prob write
__global__ void __launch_bounds__(256) k_vq(const _Float16* __restrict__ zf16,
                                            const _Float16* __restrict__ chalf,
                                            const float* __restrict__ cs,
                                            int* __restrict__ idx,
                                            float* __restrict__ prob) {
    int p = blockIdx.x / NT, nt = blockIdx.x % NT;
    int wave = threadIdx.x >> 5, lane = threadIdx.x & 31;
    int l16 = lane & 15, hi = lane >> 4;        // hi: which 16-lane half
    int nbase = nt * 128 + wave * 16;

    // A fragments: lane holds row nbase+l16; contiguous K order (k = 16*hi + s), matching B
    AF a[4];
    const _Float16* arow = zf16 + ((size_t)p * NN + (nbase + l16)) * DD;
#pragma unroll
    for (int c = 0; c < 4; ++c) {
        const _Float16* ap = arow + c * 32 + hi * 16;
        a[c].h[0] = *(const h8*)(ap);
        a[c].h[1] = *(const h8*)(ap + 8);
    }
    const float*    csp   = cs + p * KK;
    const _Float16* cbase = chalf + (size_t)p * KK * DD;

    float ssum[8], emax[8]; int bk[8];
#pragma unroll
    for (int i = 0; i < 8; ++i) { ssum[i] = 0.f; emax[i] = 0.f; bk[i] = 0; }

    // ---- pass 1: branchless accumulation of sum(e) and argmax(e) == argmin(dist)
    for (int k0 = 0; k0 < KK; k0 += 16) {
        const _Float16* brow = cbase + (size_t)(k0 + l16) * DD + hi * 16;
        v8f acc = dot128(a, brow);
        float csv = csp[k0 + l16];
        int   kc  = k0 + l16;
#pragma unroll
        for (int i = 0; i < 8; ++i) {
            float e = __builtin_amdgcn_exp2f(__builtin_fmaf(acc[i], C4_, csv));
            ssum[i] += e;
            bk[i]   = (e > emax[i]) ? kc : bk[i];
            emax[i] = fmaxf(emax[i], e);
        }
    }

    // ---- reduce across the 16 lanes of each half (columns of the row)
#pragma unroll
    for (int i = 0; i < 8; ++i) {
#pragma unroll
        for (int off = 1; off < 16; off <<= 1) {
            float s2 = __shfl_xor(ssum[i], off, 32);
            float e2 = __shfl_xor(emax[i], off, 32);
            int   k2 = __shfl_xor(bk[i],   off, 32);
            ssum[i] += s2;
            bool take = (e2 > emax[i]) || (e2 == emax[i] && k2 < bk[i]);
            bk[i]   = take ? k2 : bk[i];
            emax[i] = fmaxf(emax[i], e2);
        }
    }
    if (l16 == 0) {
#pragma unroll
        for (int i = 0; i < 8; ++i)
            idx[p * NN + nbase + hi * 8 + i] = bk[i];
    }
    float rs[8];
#pragma unroll
    for (int i = 0; i < 8; ++i) rs[i] = 1.0f / ssum[i];

    // ---- pass 2: recompute tiles, emit probabilities
    for (int k0 = 0; k0 < KK; k0 += 16) {
        const _Float16* brow = cbase + (size_t)(k0 + l16) * DD + hi * 16;
        v8f acc = dot128(a, brow);
        float csv = csp[k0 + l16];
#pragma unroll
        for (int i = 0; i < 8; ++i) {
            float pr = __builtin_amdgcn_exp2f(__builtin_fmaf(acc[i], C4_, csv)) * rs[i];
            size_t n = (size_t)(nbase + hi * 8 + i);
            prob[n * PK + p * KK + (k0 + l16)] = pr;
        }
    }
}

// ---------------- Kernel 4: zq_out (straight-through == gathered codebook) + loss partials
__global__ void k_zq(const float* __restrict__ z, const float* __restrict__ cb,
                     const int* __restrict__ idx, float* __restrict__ zq,
                     float* __restrict__ lossacc) {
    int o  = blockIdx.x * 256 + threadIdx.x;    // exact cover of B*C*H*W
    int hw = o % HWD;
    int bc = o / HWD;
    int c  = bc % CC_;
    int b  = bc / CC_;
    int p  = c >> 7, di = c & 127;
    int n  = b * HWD + hw;
    int k  = idx[p * NN + n];
    float q  = cb[((size_t)(p * KK + k)) * DD + di];
    float zf = z[o];
    zq[o] = q;
    float dsq = (q - zf) * (q - zf);
    __shared__ float red[256];
    red[threadIdx.x] = dsq;
    __syncthreads();
#pragma unroll
    for (int st = 128; st > 0; st >>= 1) {
        if (threadIdx.x < st) red[threadIdx.x] += red[threadIdx.x + st];
        __syncthreads();
    }
    if (threadIdx.x == 0) atomicAdd(lossacc, red[0]);
}

// ---------------- Kernel 5: q_loss = (1 + BETA) * mean((zq - zf)^2)
__global__ void k_fin(const float* __restrict__ lossacc, float* __restrict__ qloss) {
    *qloss = 1.25f * (*lossacc) / 6422528.0f;
}

extern "C" void kernel_launch(void* const* d_in, const int* in_sizes, int n_in,
                              void* d_out, int out_size, void* d_ws, size_t ws_size,
                              hipStream_t stream) {
    (void)in_sizes; (void)n_in; (void)out_size; (void)ws_size;
    const float* z  = (const float*)d_in[0];
    const float* cb = (const float*)d_in[1];
    float* out    = (float*)d_out;
    float* zq_out = out;                 // 6,422,528 floats
    float* qloss  = out + 6422528;       // 1 float
    float* prob   = out + 6422529;       // 12544*8192 floats

    char* ws = (char*)d_ws;
    _Float16* zf16  = (_Float16*)(ws);                                   // 12,845,056 B
    _Float16* chalf = (_Float16*)(ws + 12845056);                        //  2,097,152 B
    float*    cs    = (float*)(ws + 12845056 + 2097152);                 //     32,768 B
    int*      idx   = (int*)(ws + 12845056 + 2097152 + 32768);           //    200,704 B
    float*    lacc  = (float*)(ws + 12845056 + 2097152 + 32768 + 200704);

    hipMemsetAsync(lacc, 0, sizeof(float), stream);
    k_transpose<<<dim3(25, 4, 64), dim3(32, 8), 0, stream>>>(z, zf16);
    k_prepcb  <<<1024, 256, 0, stream>>>(cb, chalf, cs);
    k_vq      <<<PP * NT, 256, 0, stream>>>(zf16, chalf, cs, idx, prob);
    k_zq      <<<25088, 256, 0, stream>>>(z, cb, idx, zq_out, lacc);
    k_fin     <<<1, 1, 0, stream>>>(lacc, qloss);
}
